// AFNOLayer_481036337700
// MI455X (gfx1250) — matrix operand order repaired and checked by reading, not compiled, not run on hardware
//
#include <hip/hip_runtime.h>
#include <hip/hip_fp16.h>

typedef __attribute__((ext_vector_type(16))) _Float16 v16h;
typedef __attribute__((ext_vector_type(8)))  _Float16 v8h;
typedef __attribute__((ext_vector_type(8)))  float    v8f;

#define EDIM 768
#define NBT  16          // B*T
#define KMOD 12          // kept y modes
#define XKEEP 24         // kept x modes (5..28)
#define W32  0.19634954084936207f   // 2*pi/32

__device__ __forceinline__ float gelu_exact(float x){
    return 0.5f * x * (1.0f + erff(x * 0.70710678118654752f));
}

// ---------------- LayerNorm over E=768, one token per 256-thread block -------
__global__ __launch_bounds__(256) void ln_kernel(const float* __restrict__ x,
        const float* __restrict__ g, const float* __restrict__ b,
        float* __restrict__ outF, _Float16* __restrict__ outH)
{
    __shared__ float red[256];
    const int t = blockIdx.x;
    const int tid = threadIdx.x;
    const float* xr = x + (size_t)t * EDIM;
    float v0 = xr[tid], v1 = xr[tid + 256], v2 = xr[tid + 512];
    red[tid] = v0 + v1 + v2; __syncthreads();
    for (int off = 128; off > 0; off >>= 1){
        if (tid < off) red[tid] += red[tid + off];
        __syncthreads();
    }
    float mean = red[0] * (1.0f / EDIM);
    __syncthreads();
    float d0 = v0 - mean, d1 = v1 - mean, d2 = v2 - mean;
    red[tid] = d0*d0 + d1*d1 + d2*d2; __syncthreads();
    for (int off = 128; off > 0; off >>= 1){
        if (tid < off) red[tid] += red[tid + off];
        __syncthreads();
    }
    float rs = rsqrtf(red[0] * (1.0f / EDIM) + 1e-5f);
    float y0 = d0 * rs * g[tid]       + b[tid];
    float y1 = d1 * rs * g[tid + 256] + b[tid + 256];
    float y2 = d2 * rs * g[tid + 512] + b[tid + 512];
    size_t base = (size_t)t * EDIM;
    if (outF){ outF[base+tid]=y0; outF[base+tid+256]=y1; outF[base+tid+512]=y2; }
    if (outH){ outH[base+tid]=(_Float16)y0; outH[base+tid+256]=(_Float16)y1; outH[base+tid+512]=(_Float16)y2; }
}

// -------- forward DFT along y: G[bt,x,ym,e] = (1/32) * sum_y h * e^{-i*2pi*ym*y/32}
__global__ __launch_bounds__(256) void fftY_kernel(const float* __restrict__ h,
        float* __restrict__ Gr, float* __restrict__ Gi)
{
    size_t idx = (size_t)blockIdx.x * 256 + threadIdx.x;   // [bt][x][ym][e]
    int e  = (int)(idx % EDIM);
    int ym = (int)((idx / EDIM) % KMOD);
    int x  = (int)((idx / (EDIM * KMOD)) % 32);
    int bt = (int)( idx / ((size_t)EDIM * KMOD * 32));
    const float* hp = h + ((size_t)bt * 1024 + (size_t)x * 32) * EDIM + e;
    float sr = 0.f, si = 0.f;
    float w = W32 * (float)ym;
    for (int y = 0; y < 32; y++){
        float v = hp[(size_t)y * EDIM];
        float s, c; __sincosf(w * (float)y, &s, &c);
        sr += v * c; si -= v * s;
    }
    Gr[idx] = sr * 0.03125f;
    Gi[idx] = si * 0.03125f;
}

// -------- forward DFT along x, kept modes xm=5..28: F = sum_x G * e^{-i*th}
__global__ __launch_bounds__(256) void fftX_kernel(const float* __restrict__ Gr,
        const float* __restrict__ Gi, float* __restrict__ fr, float* __restrict__ fi)
{
    size_t idx = (size_t)blockIdx.x * 256 + threadIdx.x;   // [bt][xq][ym][e]
    int e  = (int)(idx % EDIM);
    int ym = (int)((idx / EDIM) % KMOD);
    int xq = (int)((idx / (EDIM * KMOD)) % XKEEP);
    int bt = (int)( idx / ((size_t)EDIM * KMOD * XKEEP));
    int xm = xq + 5;
    const size_t strideX = (size_t)KMOD * EDIM;
    const float* grp = Gr + ((size_t)bt * 32 * KMOD + ym) * EDIM + e;
    const float* gip = Gi + ((size_t)bt * 32 * KMOD + ym) * EDIM + e;
    float ar = 0.f, ai = 0.f;
    float w = W32 * (float)xm;
    for (int x = 0; x < 32; x++){
        float s, c; __sincosf(w * (float)x, &s, &c);
        float gr = grp[(size_t)x * strideX], gi = gip[(size_t)x * strideX];
        ar += gr * c + gi * s;     // (gr+i*gi)*(c - i*s)
        ai += gi * c - gr * s;
    }
    fr[idx] = ar; fi[idx] = ai;
}

// -------- block-diagonal complex MLP layer (64x64 per block), mode 0: gelu, 1: softshrink
__global__ __launch_bounds__(256) void bmlp_kernel(const float* __restrict__ xr,
        const float* __restrict__ xi, const float* __restrict__ w,
        const float* __restrict__ bias, float* __restrict__ outr,
        float* __restrict__ outi, int mode)
{
    size_t idx = (size_t)blockIdx.x * 256 + threadIdx.x;   // [pos][n][o]
    int o = (int)(idx & 63);
    int n = (int)((idx >> 6) % 12);
    size_t pos = idx / 768;
    const float* xrp = xr + pos * 768 + n * 64;
    const float* xip = xi + pos * 768 + n * 64;
    const float* wr = w + (size_t)n * 4096 + o;        // w[0][n][i][o]
    const float* wi = wr + 49152;                      // w[1][n][i][o]
    float sr = bias[n * 64 + o];
    float si = bias[768 + n * 64 + o];
    for (int i = 0; i < 64; i++){
        float a = xrp[i], bb = xip[i];
        float cr = wr[(size_t)i * 64], ci = wi[(size_t)i * 64];
        sr += a * cr - bb * ci;
        si += bb * cr + a * ci;
    }
    if (mode == 0){
        sr = gelu_exact(sr); si = gelu_exact(si);
    } else {
        sr = (sr > 0.01f) ? sr - 0.01f : ((sr < -0.01f) ? sr + 0.01f : 0.f);
        si = (si > 0.01f) ? si - 0.01f : ((si < -0.01f) ? si + 0.01f : 0.f);
    }
    outr[idx] = sr; outi[idx] = si;
}

// -------- inverse DFT along x: g[bt,x,ym,e] = sum_{xq} k * e^{+i*2pi*(xq+5)*x/32}
__global__ __launch_bounds__(256) void ifftX_kernel(const float* __restrict__ kr,
        const float* __restrict__ ki, float* __restrict__ gr, float* __restrict__ gi)
{
    size_t idx = (size_t)blockIdx.x * 256 + threadIdx.x;   // [bt][x][ym][e]
    int e  = (int)(idx % EDIM);
    int ym = (int)((idx / EDIM) % KMOD);
    int x  = (int)((idx / (EDIM * KMOD)) % 32);
    int bt = (int)( idx / ((size_t)EDIM * KMOD * 32));
    const size_t strideXq = (size_t)KMOD * EDIM;
    const float* krp = kr + ((size_t)bt * XKEEP * KMOD + ym) * EDIM + e;
    const float* kip = ki + ((size_t)bt * XKEEP * KMOD + ym) * EDIM + e;
    float ar = 0.f, ai = 0.f;
    float w = W32 * (float)x;
    for (int xq = 0; xq < XKEEP; xq++){
        float s, c; __sincosf(w * (float)(xq + 5), &s, &c);
        float rr = krp[(size_t)xq * strideXq], ii = kip[(size_t)xq * strideXq];
        ar += rr * c - ii * s;     // (rr+i*ii)*(c + i*s)
        ai += rr * s + ii * c;
    }
    gr[idx] = ar; gi[idx] = ai;
}

// -------- inverse rfft along y (Hermitian doubling) + h + residual ----------
__global__ __launch_bounds__(256) void ifftY_add_kernel(const float* __restrict__ gr,
        const float* __restrict__ gi, const float* __restrict__ h,
        const float* __restrict__ inp, float* __restrict__ res2)
{
    size_t idx = (size_t)blockIdx.x * 256 + threadIdx.x;   // [bt][x][y][e] == token*768+e
    int e = (int)(idx % EDIM);
    int y = (int)((idx / EDIM) % 32);
    int x = (int)((idx / (EDIM * 32)) % 32);
    int bt = (int)( idx / ((size_t)EDIM * 1024));
    const float* grp = gr + ((size_t)(bt * 32 + x) * KMOD) * EDIM + e;
    const float* gip = gi + ((size_t)(bt * 32 + x) * KMOD) * EDIM + e;
    float acc = grp[0];                 // ym=0: cos=1, sin=0, weight 1
    float w = W32 * (float)y;
    for (int ym = 1; ym < KMOD; ym++){
        float s, c; __sincosf(w * (float)ym, &s, &c);
        acc += 2.f * (grp[(size_t)ym * EDIM] * c - gip[(size_t)ym * EDIM] * s);
    }
    res2[idx] = acc * 0.03125f + h[idx] + inp[idx];
}

// -------- transpose + fp32->f16 convert: src[K][N] -> dst[N][K] -------------
__global__ __launch_bounds__(256) void transpose_h_kernel(const float* __restrict__ src,
        _Float16* __restrict__ dst, int K, int N)
{
    size_t idx = (size_t)blockIdx.x * 256 + threadIdx.x;
    if (idx >= (size_t)K * N) return;
    int k = (int)(idx % K);
    int n = (int)(idx / K);
    dst[idx] = (_Float16)src[(size_t)k * N + n];
}

// -------- WMMA GEMM with register tiling ------------------------------------
// Each wave: 16(M) x 64(N) strip = 4 accumulators sharing one A fragment.
// Block: 8 waves as 2(M) x 4(N) -> 32 x 256 output tile.
// A: [M][K] f16 row-major.  BT: [N][K] f16 (pre-transposed weights).
// do_gelu=1: outH = f16(gelu(acc+bias)).  do_gelu=0: outF = acc+bias+res.
__global__ __launch_bounds__(256) void wmma_gemm_kernel(
    const _Float16* __restrict__ A, const _Float16* __restrict__ BT,
    const float* __restrict__ bias, const float* __restrict__ res,
    float* __restrict__ outF, _Float16* __restrict__ outH,
    int N, int K, int do_gelu)
{
    const int wave = threadIdx.x >> 5;       // 0..7
    const int lane = threadIdx.x & 31;
    const int g = lane >> 4;                 // half-wave group
    const int r = lane & 15;                 // A: M row / B: N col (within tile)
    const int m0 = blockIdx.y * 32 + (wave >> 2) * 16;
    const int n0 = blockIdx.x * 256 + (wave & 3) * 64;
    const _Float16* Arow = A  + (size_t)(m0 + r) * K;
    const _Float16* B0   = BT + (size_t)(n0 + r) * K;
    const size_t bstride = (size_t)16 * K;   // 16 N-rows between sub-tiles
    v8f acc[4] = {};
    for (int k = 0; k < K; k += 32){
        __builtin_prefetch(Arow + k + 512, 0, 1);
        __builtin_prefetch(B0   + k + 512, 0, 1);
        // A frag: elements 0..7 -> K = k + g*8 + i ; 8..15 -> +16
        v8h a0 = *(const v8h*)(Arow + k + g * 8);
        v8h a1 = *(const v8h*)(Arow + k + 16 + g * 8);
        v16h a = __builtin_shufflevector(a0, a1, 0,1,2,3,4,5,6,7,8,9,10,11,12,13,14,15);
        #pragma unroll
        for (int j = 0; j < 4; j++){
            // B frag: elements 0..7 -> K = k + g*8 + i ; 8..15 -> +16
            const _Float16* Bp = B0 + (size_t)j * bstride + k + g * 8;
            v8h b0 = *(const v8h*)(Bp);
            v8h b1 = *(const v8h*)(Bp + 16);
            v16h b = __builtin_shufflevector(b0, b1, 0,1,2,3,4,5,6,7,8,9,10,11,12,13,14,15);
            acc[j] = __builtin_amdgcn_wmma_f32_16x16x32_f16(
                        false, a, false, b, (short)0, acc[j], false, false);
        }
    }
    #pragma unroll
    for (int j = 0; j < 4; j++){
        const int n = n0 + 16 * j + r;
        float bn = bias[n];
        #pragma unroll
        for (int v = 0; v < 8; v++){
            int m = m0 + v + 8 * g;          // D layout: VGPR v -> row v + 8*(lane/16)
            size_t o = (size_t)m * N + n;
            float val = acc[j][v] + bn;
            if (do_gelu) outH[o] = (_Float16)gelu_exact(val);
            else         outF[o] = val + res[o];
        }
    }
}

extern "C" void kernel_launch(void* const* d_in, const int* in_sizes, int n_in,
                              void* d_out, int out_size, void* d_ws, size_t ws_size,
                              hipStream_t stream) {
    const float* input = (const float*)d_in[0];
    const float* w1    = (const float*)d_in[1];
    const float* b1    = (const float*)d_in[2];
    const float* w2    = (const float*)d_in[3];
    const float* b2    = (const float*)d_in[4];
    const float* ln1g  = (const float*)d_in[5];
    const float* ln1b  = (const float*)d_in[6];
    const float* ln2g  = (const float*)d_in[7];
    const float* ln2b  = (const float*)d_in[8];
    const float* mw1   = (const float*)d_in[9];
    const float* mb1   = (const float*)d_in[10];
    const float* mw2   = (const float*)d_in[11];
    const float* mb2   = (const float*)d_in[12];

    const size_t NTOK = 16384;               // B*T*NX*NY
    const size_t G_ELEMS = (size_t)NBT * 32 * KMOD * EDIM;       // 4,718,592
    const size_t F_ELEMS = (size_t)NBT * XKEEP * KMOD * EDIM;    // 3,538,944

    char* p = (char*)d_ws;
    float*    hbuf = (float*)p;    p += NTOK * EDIM * 4;
    float*    Gr   = (float*)p;    p += G_ELEMS * 4;
    float*    Gi   = (float*)p;    p += G_ELEMS * 4;
    float*    fr   = (float*)p;    p += F_ELEMS * 4;
    float*    fi   = (float*)p;    p += F_ELEMS * 4;
    float*    o1r  = (float*)p;    p += F_ELEMS * 4;
    float*    o1i  = (float*)p;    p += F_ELEMS * 4;
    float*    res2 = (float*)p;    p += NTOK * EDIM * 4;
    _Float16* xln  = (_Float16*)p; p += NTOK * EDIM * 2;
    _Float16* mid  = (_Float16*)p; p += NTOK * 3072 * 2;
    _Float16* w1t  = (_Float16*)p; p += (size_t)3072 * EDIM * 2;
    _Float16* w2t  = (_Float16*)p; p += (size_t)EDIM * 3072 * 2;

    // 1) LN1 -> h
    ln_kernel<<<16384, 256, 0, stream>>>(input, ln1g, ln1b, hbuf, nullptr);
    // 2) forward DFT y (12 kept modes)
    fftY_kernel<<<(int)(G_ELEMS / 256), 256, 0, stream>>>(hbuf, Gr, Gi);
    // 3) forward DFT x (modes 5..28)
    fftX_kernel<<<(int)(F_ELEMS / 256), 256, 0, stream>>>(Gr, Gi, fr, fi);
    // 4) block MLP layer 1 (+gelu)
    bmlp_kernel<<<(int)(F_ELEMS / 256), 256, 0, stream>>>(fr, fi, w1, b1, o1r, o1i, 0);
    // 5) block MLP layer 2 (+softshrink), overwrite fr/fi with kept-mode output
    bmlp_kernel<<<(int)(F_ELEMS / 256), 256, 0, stream>>>(o1r, o1i, w2, b2, fr, fi, 1);
    // 6) inverse DFT x
    ifftX_kernel<<<(int)(G_ELEMS / 256), 256, 0, stream>>>(fr, fi, Gr, Gi);
    // 7) inverse rfft y + h + residual -> res2
    ifftY_add_kernel<<<(int)(NTOK * EDIM / 256), 256, 0, stream>>>(Gr, Gi, hbuf, input, res2);
    // 8) LN2 -> f16
    ln_kernel<<<16384, 256, 0, stream>>>(res2, ln2g, ln2b, nullptr, xln);
    // 9) weight prep: mw1[768][3072] -> w1t[3072][768]; mw2[3072][768] -> w2t[768][3072]
    transpose_h_kernel<<<9216, 256, 0, stream>>>(mw1, w1t, 768, 3072);
    transpose_h_kernel<<<9216, 256, 0, stream>>>(mw2, w2t, 3072, 768);
    // 10) GEMM1: mid = gelu(xln @ mw1 + mb1)  [16384 x 3072], f16 out
    wmma_gemm_kernel<<<dim3(3072 / 256, 16384 / 32), 256, 0, stream>>>(
        xln, w1t, mb1, nullptr, nullptr, mid, 3072, 768, 1);
    // 11) GEMM2: out = mid @ mw2 + mb2 + res2  [16384 x 768], fp32 out
    wmma_gemm_kernel<<<dim3(768 / 256, 16384 / 32), 256, 0, stream>>>(
        mid, w2t, mb2, res2, (float*)d_out, nullptr, 768, 3072, 0);
}
